// NonRedundantMultiheadAttention_23733989278387
// MI455X (gfx1250) — matrix operand order repaired and checked by reading, not compiled, not run on hardware
//
#include <hip/hip_runtime.h>
#include <hip/hip_bf16.h>

typedef __attribute__((ext_vector_type(16))) _Float16 v16h;
typedef __attribute__((ext_vector_type(8)))  float    v8f;

#define HD   64
#define NH   16
#define NB   2
#define NSEQ 2048
#define EMB  1024
#define MROWS (NB*NSEQ)   // 4096

union Frag16 { v16h v; _Float16 h[16]; unsigned u[8]; };

__device__ inline v8f wmma_f16(v16h a, v16h b, v8f c) {
  // D = A(16x32 f16) * B(32x16 f16) + C(16x16 f32)
  return __builtin_amdgcn_wmma_f32_16x16x32_f16(false, a, false, b, (short)0, c,
                                                false, false);
}

// CDNA5 async Global->LDS DMA: one 16-byte transfer per active lane,
// tracked by ASYNCcnt (CDNA5 ISA).
__device__ inline void async_b128(unsigned lds_off, unsigned long long gaddr) {
  asm volatile("global_load_async_to_lds_b128 %0, %1, off"
               :: "v"(lds_off), "v"(gaddr) : "memory");
}
__device__ inline void wait_async0() {
  asm volatile("s_wait_asynccnt 0x0" ::: "memory");
}
__device__ inline unsigned lds_off_of(const void* p) {
  // LDS aperture: addr[31:0] is the LDS byte offset (ISA 10.2 aperture calc)
  return (unsigned)(unsigned long long)(uintptr_t)p;
}

// ---------------------------------------------------------------- conversions
__global__ void cvt_f32_f16(const float* __restrict__ in,
                            _Float16* __restrict__ out, int n) {
  for (int i = blockIdx.x * blockDim.x + threadIdx.x; i < n;
       i += gridDim.x * blockDim.x)
    out[i] = (_Float16)in[i];
}

// ---------------------------------------------------------------- GEMM
// Y(M x 1024) = X(M x 1024,f16) @ W(1024 x 1024,f16 row=out-col)^T, then
// (acc + bias) * scale.  grid=(M/128, 1024/64), block=256 (8 waves).
// Wave w computes rows [m0+w*16, +16) x cols [n0, n0+64) as 4 WMMA tiles.
// W tile (64 rows x 32 k = 4KB) staged in LDS by async DMA, double buffered.
// out16 modes: vtrans=0 -> (B,H,N,64) head-major; vtrans=1 -> (B,H,64,N)
// d-major (used for V so the attention P@V B-fragments are contiguous).
__global__ void gemm_wmma_kernel(const _Float16* __restrict__ X,
                                 const _Float16* __restrict__ W,
                                 const float* __restrict__ bias, float scale,
                                 _Float16* __restrict__ out16, int vtrans,
                                 float* __restrict__ out32) {
  __shared__ _Float16 Bt[2][64][32];   // [buf][n_local][k_local]

  const int tid = threadIdx.x;
  const int lane = tid & 31;
  const int wave = tid >> 5;
  const int m0 = blockIdx.x * 128 + wave * 16;
  const int n0 = blockIdx.y * 64;
  const int K = EMB;

  const unsigned* __restrict__ Xu = (const unsigned*)X;

  // async staging: thread t moves 16B of row (t>>2), segment (t&3) of the
  // 64B row chunk W[n0+row][k0 .. k0+32)
  const int strow = tid >> 2;          // 0..63
  const int stseg = tid & 3;           // 0..3
  const unsigned long long wrow_gbase =
      (unsigned long long)(uintptr_t)(W + (size_t)(n0 + strow) * K) +
      (unsigned)stseg * 16u;
  const unsigned bt_lds[2] = {
      lds_off_of(&Bt[0][strow][0]) + (unsigned)stseg * 16u,
      lds_off_of(&Bt[1][strow][0]) + (unsigned)stseg * 16u};

  const int arow = m0 + (lane & 15);
  // A k order chosen here (contiguous, B matches): k = (lane>=16?16:0) + v*2
  int akoff[8];
#pragma unroll
  for (int v2 = 0; v2 < 8; ++v2)
    akoff[v2] = ((lane >> 4) ? 16 : 0) + v2 * 2;

  v8f vzero = {0.f, 0.f, 0.f, 0.f, 0.f, 0.f, 0.f, 0.f};
  v8f acc[4];
#pragma unroll
  for (int t = 0; t < 4; ++t) acc[t] = vzero;

  async_b128(bt_lds[0], wrow_gbase);   // prologue: stage k0 = 0

  for (int k0 = 0; k0 < K; k0 += 32) {
    const int cur = (k0 >> 5) & 1;
    wait_async0();        // own transfers into Bt[cur] complete
    __syncthreads();      // whole WG's transfers done; prev reads done
    if (k0 + 32 < K)      // prefetch next W tile into the other buffer
      async_b128(bt_lds[cur ^ 1], wrow_gbase + (unsigned)(k0 + 32) * 2u);

    Frag16 a;
#pragma unroll
    for (int v2 = 0; v2 < 8; ++v2)
      a.u[v2] = Xu[((size_t)arow * K + (k0 + akoff[v2])) >> 1];

    // hoist all 4 B fragments -> one DS clause + single wait, then 4 WMMAs
    const int kb2 = (lane >> 4) ? 16 : 0;
    Frag16 bf[4];
#pragma unroll
    for (int nt = 0; nt < 4; ++nt) {
      const int nloc = nt * 16 + (lane & 15);
#pragma unroll
      for (int v2 = 0; v2 < 8; ++v2)
        bf[nt].u[v2] = *(const unsigned*)&Bt[cur][nloc][kb2 + v2 * 2];
    }
#pragma unroll
    for (int nt = 0; nt < 4; ++nt)
      acc[nt] = wmma_f16(a.v, bf[nt].v, acc[nt]);
  }

  // C-layout: row = r + (lane>=16?8:0), col-in-tile = lane&15
  if (out16) {
#pragma unroll
    for (int nt = 0; nt < 4; ++nt) {
      const int col = n0 + nt * 16 + (lane & 15);
      const float bv = bias[col];
      const int hh = col >> 6, dd = col & 63;
#pragma unroll
      for (int r = 0; r < 8; ++r) {
        const int row = m0 + r + ((lane >> 4) ? 8 : 0);
        const int bb = row >> 11, nn = row & (NSEQ - 1);
        const float val = (acc[nt][r] + bv) * scale;
        const size_t idx =
            vtrans ? (((size_t)bb * NH + hh) * HD + dd) * NSEQ + nn
                   : (((size_t)bb * NH + hh) * NSEQ + nn) * HD + dd;
        out16[idx] = (_Float16)val;
      }
    }
  } else {
#pragma unroll
    for (int nt = 0; nt < 4; ++nt) {
      const int col = n0 + nt * 16 + (lane & 15);
      const float bv = bias[col];
#pragma unroll
      for (int r = 0; r < 8; ++r) {
        const int row = m0 + r + ((lane >> 4) ? 8 : 0);
        out32[(size_t)row * EMB + col] = (acc[nt][r] + bv) * scale;
      }
    }
  }
}

// ---------------------------------------------------------------- attention
// Flash attention per (b,h). Q/K in (B,H,N,64); V in (B,H,64,N) f16.
// Scale folded into Q; per-key bias log(abundance) added to scores.
// grid=(N/128, B*H), block=256 (8 waves); wave owns 16 query rows.
// K (32x64 row-major) and V (64x32 d-major) blocks staged by async DMA,
// double buffered; all WMMA fragments are aligned ds_load_b128 pairs.
__global__ void attn_wmma_kernel(const _Float16* __restrict__ Q,
                                 const _Float16* __restrict__ Kx,
                                 const _Float16* __restrict__ V,
                                 const float* __restrict__ abund,
                                 _Float16* __restrict__ O) {
  __shared__ _Float16 Kt[2][32][64];   // [buf][key][d]
  __shared__ _Float16 Vt[2][64][32];   // [buf][d][key]   (d-major)
  __shared__ _Float16 Pt[8][16][32];   // per-wave P staging (C->A relayout)

  const int tid = threadIdx.x;
  const int lane = tid & 31;
  const int wave = tid >> 5;
  const int bh = blockIdx.y;           // = b*NH + h
  const int bb = bh >> 4;
  const int hh = bh & 15;
  const int q0 = blockIdx.x * 128 + wave * 16;
  const size_t base = (size_t)bh * NSEQ * HD;

  // Load per-wave Q fragments once: rows q0..q0+15, d split 0..31 / 32..63.
  Frag16 aq0, aq1;
  {
    const unsigned* Qu = (const unsigned*)(Q + base);
    const int row = q0 + (lane & 15);
#pragma unroll
    for (int v2 = 0; v2 < 8; ++v2) {
      const int k = ((lane >> 4) ? 16 : 0) + v2 * 2;
      aq0.u[v2] = Qu[((size_t)row * HD + k) >> 1];
      aq1.u[v2] = Qu[((size_t)row * HD + 32 + k) >> 1];
    }
  }

  float mrow[8], lrow[8];
#pragma unroll
  for (int r = 0; r < 8; ++r) { mrow[r] = -1e30f; lrow[r] = 0.f; }
  v8f vzero = {0.f, 0.f, 0.f, 0.f, 0.f, 0.f, 0.f, 0.f};
  v8f acc[4];
#pragma unroll
  for (int t = 0; t < 4; ++t) acc[t] = vzero;

  // K staging: flat 4KB copy (rows contiguous), thread t moves bytes t*16.
  const unsigned long long kg =
      (unsigned long long)(uintptr_t)(Kx + base) + (unsigned)tid * 16u;
  // V staging (d-major source): thread t moves 16B seg (t&3) of d-row (t>>2);
  // global row stride = NSEQ*2 bytes; per-block offset = j0*2 bytes.
  const unsigned long long vg =
      (unsigned long long)(uintptr_t)(V + base + (size_t)(tid >> 2) * NSEQ) +
      (unsigned)(tid & 3) * 16u;
  const unsigned kt_lds[2] = {lds_off_of(&Kt[0][0][0]) + (unsigned)tid * 16u,
                              lds_off_of(&Kt[1][0][0]) + (unsigned)tid * 16u};
  const unsigned vt_lds[2] = {
      lds_off_of(&Vt[0][tid >> 2][0]) + (unsigned)(tid & 3) * 16u,
      lds_off_of(&Vt[1][tid >> 2][0]) + (unsigned)(tid & 3) * 16u};
  const float* abase = abund + (size_t)bb * NSEQ;

  // prologue: stage key block j0 = 0 into buffer 0
  async_b128(kt_lds[0], kg);
  async_b128(vt_lds[0], vg);

  for (int j0 = 0; j0 < NSEQ; j0 += 32) {
    const int cur = (j0 >> 5) & 1;
    wait_async0();        // own transfers into buf[cur] complete
    __syncthreads();      // WG transfers done; prev reads of buf[cur^1] done
    if (j0 + 32 < NSEQ) { // prefetch next K/V block
      async_b128(kt_lds[cur ^ 1], kg + (unsigned)(j0 + 32) * (HD * 2));
      async_b128(vt_lds[cur ^ 1], vg + (unsigned)(j0 + 32) * 2u);
    }

    // S tiles: hoist 4 B fragments, then 4 WMMAs back-to-back.
    // B[d][key]: lane = key col (kk0/kk1), d striped 2-per-VGPR.
    Frag16 b00, b01, b10, b11;
    {
      const int kk0 = lane & 15;
      const int kk1 = 16 + (lane & 15);
      const int db = (lane >> 4) ? 16 : 0;
#pragma unroll
      for (int v2 = 0; v2 < 8; ++v2) {
        b00.u[v2] = *(const unsigned*)&Kt[cur][kk0][db + v2 * 2];
        b01.u[v2] = *(const unsigned*)&Kt[cur][kk1][db + v2 * 2];
        b10.u[v2] = *(const unsigned*)&Kt[cur][kk0][32 + db + v2 * 2];
        b11.u[v2] = *(const unsigned*)&Kt[cur][kk1][32 + db + v2 * 2];
      }
    }
    v8f s0 = vzero, s1 = vzero;
    s0 = wmma_f16(aq0.v, b00.v, s0);
    s1 = wmma_f16(aq0.v, b01.v, s1);
    s0 = wmma_f16(aq1.v, b10.v, s0);
    s1 = wmma_f16(aq1.v, b11.v, s1);

    // Online softmax update. C-row r lives in one 16-lane group -> xor-shfl.
    const float lb0 = __logf(abase[j0 + (lane & 15)]);
    const float lb1 = __logf(abase[j0 + 16 + (lane & 15)]);
#pragma unroll
    for (int r = 0; r < 8; ++r) {
      const float x0 = s0[r] + lb0;
      const float x1 = s1[r] + lb1;
      float mt = fmaxf(x0, x1);
#pragma unroll
      for (int w = 1; w < 16; w <<= 1) mt = fmaxf(mt, __shfl_xor(mt, w, 32));
      const float mnew = fmaxf(mrow[r], mt);
      const float corr = __expf(mrow[r] - mnew);
      const float p0 = __expf(x0 - mnew);
      const float p1 = __expf(x1 - mnew);
      float rs = p0 + p1;
#pragma unroll
      for (int w = 1; w < 16; w <<= 1) rs += __shfl_xor(rs, w, 32);
      lrow[r] = lrow[r] * corr + rs;
      mrow[r] = mnew;
#pragma unroll
      for (int t = 0; t < 4; ++t) acc[t][r] *= corr;
      // stage P (C layout -> LDS row-major, per-wave buffer; no barrier)
      const int prow = r + ((lane >> 4) ? 8 : 0);
      Pt[wave][prow][lane & 15] = (_Float16)p0;
      Pt[wave][prow][16 + (lane & 15)] = (_Float16)p1;
    }

    // P(16x32) @ V(32x64): A from Pt; B from d-major Vt (contiguous b128s).
    Frag16 ap;
#pragma unroll
    for (int v2 = 0; v2 < 8; ++v2) {
      const int k = ((lane >> 4) ? 16 : 0) + v2 * 2;
      ap.u[v2] = *(const unsigned*)&Pt[wave][lane & 15][k];
    }
    Frag16 bv[4];
    {
      const int kb = (lane >> 4) ? 16 : 0;
#pragma unroll
      for (int t = 0; t < 4; ++t) {
        const int d = t * 16 + (lane & 15);
#pragma unroll
        for (int v2 = 0; v2 < 8; ++v2)
          bv[t].u[v2] = *(const unsigned*)&Vt[cur][d][kb + v2 * 2];
      }
    }
#pragma unroll
    for (int t = 0; t < 4; ++t)
      acc[t] = wmma_f16(ap.v, bv[t].v, acc[t]);
  }

  // Epilogue: O[b, n, h*64+d] = acc / l  (f16, row-major 4096x1024)
#pragma unroll
  for (int t = 0; t < 4; ++t) {
#pragma unroll
    for (int r = 0; r < 8; ++r) {
      const int row = q0 + r + ((lane >> 4) ? 8 : 0);
      const int d = t * 16 + (lane & 15);
      const float val = acc[t][r] / lrow[r];
      O[((size_t)bb * NSEQ + row) * EMB + hh * HD + d] = (_Float16)val;
    }
  }
}

// ---------------------------------------------------------------- launch
extern "C" void kernel_launch(void* const* d_in, const int* in_sizes, int n_in,
                              void* d_out, int out_size, void* d_ws,
                              size_t ws_size, hipStream_t stream) {
  (void)in_sizes; (void)n_in; (void)out_size; (void)ws_size;
  const float* query = (const float*)d_in[0];
  const float* key   = (const float*)d_in[1];
  const float* value = (const float*)d_in[2];
  const float* abund = (const float*)d_in[3];
  const float* win   = (const float*)d_in[4];  // (3*1024, 1024)
  const float* bin   = (const float*)d_in[5];  // (3*1024,)
  const float* wout  = (const float*)d_in[6];  // (1024, 1024)
  const float* bout  = (const float*)d_in[7];  // (1024,)
  float* out = (float*)d_out;

  char* ws = (char*)d_ws;
  size_t off = 0;
  auto alloc = [&](size_t bytes) -> void* {
    void* p = ws + off;
    off += (bytes + 255) & ~(size_t)255;
    return p;
  };
  _Float16* x16q   = (_Float16*)alloc((size_t)MROWS * EMB * 2);
  _Float16* x16k   = (_Float16*)alloc((size_t)MROWS * EMB * 2);
  _Float16* x16v   = (_Float16*)alloc((size_t)MROWS * EMB * 2);
  _Float16* w16in  = (_Float16*)alloc((size_t)3 * EMB * EMB * 2);
  _Float16* w16out = (_Float16*)alloc((size_t)EMB * EMB * 2);
  _Float16* qproj  = (_Float16*)alloc((size_t)MROWS * EMB * 2);
  _Float16* kproj  = (_Float16*)alloc((size_t)MROWS * EMB * 2);
  _Float16* vproj  = (_Float16*)alloc((size_t)MROWS * EMB * 2);
  _Float16* attno  = (_Float16*)alloc((size_t)MROWS * EMB * 2);

  cvt_f32_f16<<<1024, 256, 0, stream>>>(query, x16q, MROWS * EMB);
  cvt_f32_f16<<<1024, 256, 0, stream>>>(key,   x16k, MROWS * EMB);
  cvt_f32_f16<<<1024, 256, 0, stream>>>(value, x16v, MROWS * EMB);
  cvt_f32_f16<<<1024, 256, 0, stream>>>(win,   w16in, 3 * EMB * EMB);
  cvt_f32_f16<<<1024, 256, 0, stream>>>(wout,  w16out, EMB * EMB);

  dim3 ggrid(MROWS / 128, EMB / 64);   // (32, 16)
  const float qscale = 0.125f;         // 1/sqrt(64), folded into Q projection
  gemm_wmma_kernel<<<ggrid, 256, 0, stream>>>(x16q, w16in, bin, qscale,
                                              qproj, 0, nullptr);
  gemm_wmma_kernel<<<ggrid, 256, 0, stream>>>(x16k, w16in + (size_t)EMB * EMB,
                                              bin + EMB, 1.0f, kproj, 0,
                                              nullptr);
  gemm_wmma_kernel<<<ggrid, 256, 0, stream>>>(x16v, w16in + (size_t)2 * EMB * EMB,
                                              bin + 2 * EMB, 1.0f, vproj, 1,
                                              nullptr);

  dim3 agrid(NSEQ / 128, NB * NH);     // (16, 32)
  attn_wmma_kernel<<<agrid, 256, 0, stream>>>(qproj, kproj, vproj, abund, attno);

  // Final out-projection writes the f32 result to d_out.
  gemm_wmma_kernel<<<ggrid, 256, 0, stream>>>(attno, w16out, bout, 1.0f,
                                              nullptr, 0, out);
}
